// ORCDF_EX_55276229099953
// MI455X (gfx1250) — matrix operand order repaired and checked by reading, not compiled, not run on hardware
//
#include <hip/hip_runtime.h>
#include <hip/hip_bf16.h>
#include <math.h>

// Problem constants (from reference)
#define S_    10000
#define EX_   5000
#define KK_   500
#define DD_   128
#define NN_   15500          // S + EX + K
#define LL_   3
#define NNZ_  500000
#define BB_   8192
#define ND_   (NN_ * DD_)
#define LDT_  15504          // transposed leading dim, padded to 16 elems (32B rows)
#define KPAD_ 512            // projection weight cols padded 500 -> 512
#define SSL_W 0.1f
#define INV_T 2.0f           // 1 / 0.5
#define JS_   4              // j-sweep split factor for info_nce

typedef __attribute__((ext_vector_type(16))) __bf16 v16bf;
typedef __attribute__((ext_vector_type(8)))  float  v8f;

__device__ __forceinline__ __bf16 f2bf(float f) { return (__bf16)f; }

// ---------------------------------------------------------------------------
// all_emb = concat(stu, exer, know) rows
// ---------------------------------------------------------------------------
__global__ void k_init_all(const float* __restrict__ stu,
                           const float* __restrict__ exer,
                           const float* __restrict__ know,
                           float* __restrict__ all0) {
  int i = blockIdx.x * blockDim.x + threadIdx.x;
  if (i >= ND_) return;
  int row = i / DD_, col = i - row * DD_;
  float v;
  if (row < S_)            v = stu[(long)row * DD_ + col];
  else if (row < S_ + EX_) v = exer[(long)(row - S_) * DD_ + col];
  else                     v = know[(long)(row - S_ - EX_) * DD_ + col];
  all0[i] = v;
}

__global__ void k_zero(float* __restrict__ p, int n) {
  int i = blockIdx.x * blockDim.x + threadIdx.x;
  if (i < n) p[i] = 0.f;
}

__global__ void k_zero1(float* p) { if (threadIdx.x == 0 && blockIdx.x == 0) *p = 0.f; }

__global__ void k_scale(const float* __restrict__ a, float* __restrict__ o,
                        float s, int n) {
  int i = blockIdx.x * blockDim.x + threadIdx.x;
  if (i < n) o[i] = a[i] * s;
}

// f32 -> bf16 straight copy
__global__ void k_cvt_bf16(const float* __restrict__ src, __bf16* __restrict__ dst,
                           int n) {
  int i = blockIdx.x * blockDim.x + threadIdx.x;
  if (i < n) dst[i] = f2bf(src[i]);
}

// f32 [N x D] row-major -> bf16 [D x LDT_] transposed (padding cols unwritten)
__global__ void k_transpose_bf16(const float* __restrict__ src,
                                 __bf16* __restrict__ dst) {
  int i = blockIdx.x * blockDim.x + threadIdx.x;
  if (i >= ND_) return;
  int row = i / DD_, d = i - row * DD_;
  dst[(long)d * LDT_ + row] = f2bf(src[i]);
}

// f32 [128 x Kout] -> bf16 [128 x KPAD_] zero-padded
__global__ void k_pad_w(const float* __restrict__ W, __bf16* __restrict__ dst,
                        int Kout) {
  int i = blockIdx.x * blockDim.x + threadIdx.x;
  if (i >= DD_ * KPAD_) return;
  int k = i >> 9, c = i & (KPAD_ - 1);
  dst[i] = (c < Kout) ? f2bf(W[(long)k * Kout + c]) : f2bf(0.f);
}

// ---------------------------------------------------------------------------
// spMM scatter: y[rows[e], :] += vals[e] * x[cols[e], :]
// float4 gather + 4 native f32 atomics per (nnz, 4-col) thread.
// ---------------------------------------------------------------------------
__global__ void k_spmm(const int* __restrict__ rows, const int* __restrict__ cols,
                       const float* __restrict__ vals,
                       const float* __restrict__ x, float* __restrict__ y) {
  long t = (long)blockIdx.x * blockDim.x + threadIdx.x;
  long e = t >> 5;
  if (e >= NNZ_) return;
  int c4 = (int)(t & 31) * 4;
  int r = rows[e], c = cols[e];
  float v = vals[e];
  float4 xv = *(const float4*)(x + (long)c * DD_ + c4);
  float* yp = y + (long)r * DD_ + c4;
  unsafeAtomicAdd(yp + 0, v * xv.x);
  unsafeAtomicAdd(yp + 1, v * xv.y);
  unsafeAtomicAdd(yp + 2, v * xv.z);
  unsafeAtomicAdd(yp + 3, v * xv.w);
}

// ---------------------------------------------------------------------------
// acc += ([r | w] @ Wc + bc). One wave per 16-row tile, K=256 in 8 bf16 WMMA
// steps, 8 col-tiles. B fragments preloaded as a clause so WMMA chain overlaps
// load latency; single tile-level bounds test for the store path.
// ---------------------------------------------------------------------------
__global__ void k_concat_gemm_acc(const float* __restrict__ r,
                                  const float* __restrict__ w,
                                  const __bf16* __restrict__ Wcb,
                                  const float* __restrict__ bc,
                                  float* __restrict__ acc) {
  int lane = threadIdx.x & 31;
  int wave = (blockIdx.x * blockDim.x + threadIdx.x) >> 5;
  int mT = (NN_ + 15) / 16;
  if (wave >= mT) return;
  int m0 = wave * 16;
  int mr = m0 + (lane & 15);
  int mc = mr < NN_ ? mr : NN_ - 1;
  int kh = (lane >> 4) * 16;
  bool full = (m0 + 16) <= NN_;

  v16bf a[8];
#pragma unroll
  for (int ks = 0; ks < 8; ++ks) {
    int kb = ks * 32 + kh;
#pragma unroll
    for (int t = 0; t < 16; ++t) {
      int k = kb + t;
      float f = (k < DD_) ? r[(long)mc * DD_ + k]
                          : w[(long)mc * DD_ + (k - DD_)];
      a[ks][t] = f2bf(f);
    }
  }

  for (int nt = 0; nt < 8; ++nt) {
    int n0 = nt * 16;
    v16bf b[8];
#pragma unroll
    for (int ks = 0; ks < 8; ++ks)
#pragma unroll
      for (int t = 0; t < 16; ++t)
        b[ks][t] = Wcb[(long)(ks * 32 + kh + t) * DD_ + n0 + (lane & 15)];
    v8f c = {};
#pragma unroll
    for (int ks = 0; ks < 8; ++ks)
      c = __builtin_amdgcn_wmma_f32_16x16x32_bf16(false, a[ks], false, b[ks],
                                                  (short)0, c, false, false);
    int col = n0 + (lane & 15);
    int rbase = m0 + 8 * (lane >> 4);
    float bias = bc[col];
    float* ap = acc + (long)rbase * DD_ + col;
    if (full) {
#pragma unroll
      for (int v = 0; v < 8; ++v) ap[(long)v * DD_] += c[v] + bias;
    } else {
#pragma unroll
      for (int v = 0; v < 8; ++v)
        if (rbase + v < NN_) ap[(long)v * DD_] += c[v] + bias;
    }
  }
}

// ---------------------------------------------------------------------------
// info_nce partial logsumexp. One wave owns a 16-row tile of v1 and one of
// JS_ j-chunks of v2 (bf16 transposed, leading dim LDT_). B tiles double-
// buffered across j iterations; lane-local online (m,s); single 16-lane merge.
// ---------------------------------------------------------------------------
__global__ void k_nce(const __bf16* __restrict__ v1b, const __bf16* __restrict__ v2t,
                      int n, int chunkT,
                      float* __restrict__ mPart, float* __restrict__ sPart) {
  int lane = threadIdx.x & 31;
  int wave = (blockIdx.x * blockDim.x + threadIdx.x) >> 5;
  int mT = (n + 15) / 16;
  if (wave >= mT * JS_) return;
  int iTile = wave / JS_;
  int chunk = wave - iTile * JS_;
  int i0 = iTile * 16;
  int mr = i0 + (lane & 15);
  if (mr >= n) mr = n - 1;
  int kh = (lane >> 4) * 16;

  v16bf a[4];
#pragma unroll
  for (int ks = 0; ks < 4; ++ks)
    a[ks] = *(const v16bf*)(v1b + (long)mr * DD_ + ks * 32 + kh);

  float m[8], s[8];
#pragma unroll
  for (int v = 0; v < 8; ++v) { m[v] = -3.0e38f; s[v] = 0.f; }

  int jT = (n + 15) / 16;
  int jb = chunk * chunkT;
  int je = jb + chunkT; if (je > jT) je = jT;

  if (jb < je) {
    const __bf16* bBase = v2t + (long)kh * LDT_ + (lane & 15);
    v16bf b[4], bn[4];
#pragma unroll
    for (int ks = 0; ks < 4; ++ks)
#pragma unroll
      for (int t = 0; t < 16; ++t)
        b[ks][t] = bBase[(long)(ks * 32 + t) * LDT_ + jb * 16];

    for (int jt = jb; jt < je; ++jt) {
      int j0 = jt * 16;
      if (jt + 1 < je) {  // issue next tile's loads before consuming current
#pragma unroll
        for (int ks = 0; ks < 4; ++ks)
#pragma unroll
          for (int t = 0; t < 16; ++t)
            bn[ks][t] = bBase[(long)(ks * 32 + t) * LDT_ + j0 + 16];
      }
      v8f c = {};
#pragma unroll
      for (int ks = 0; ks < 4; ++ks)
        c = __builtin_amdgcn_wmma_f32_16x16x32_bf16(false, a[ks], false, b[ks],
                                                    (short)0, c, false, false);
      bool jok = (j0 + (lane & 15)) < n;
#pragma unroll
      for (int v = 0; v < 8; ++v) {
        float x = jok ? c[v] * INV_T : -3.0e38f;
        float nm = fmaxf(m[v], x);
        float e = jok ? __expf(x - nm) : 0.f;
        s[v] = s[v] * __expf(m[v] - nm) + e;
        m[v] = nm;
      }
#pragma unroll
      for (int ks = 0; ks < 4; ++ks) b[ks] = bn[ks];
    }
  }

  // merge (m,s) across the 16 lanes holding each row
#pragma unroll
  for (int msk = 1; msk < 16; msk <<= 1) {
#pragma unroll
    for (int v = 0; v < 8; ++v) {
      float om = __shfl_xor(m[v], msk, 32);
      float os = __shfl_xor(s[v], msk, 32);
      float nm = fmaxf(m[v], om);
      s[v] = s[v] * __expf(m[v] - nm) + os * __expf(om - nm);
      m[v] = nm;
    }
  }

  if ((lane & 15) == 0) {
    int base = i0 + 8 * (lane >> 4);
#pragma unroll
    for (int v = 0; v < 8; ++v) {
      int ri = base + v;
      if (ri < n) {
        mPart[(long)chunk * n + ri] = m[v];
        sPart[(long)chunk * n + ri] = s[v];
      }
    }
  }
}

// merge JS_ partials + diagonal term; accumulate loss atomically
__global__ void k_nce_fin(const float* __restrict__ v1, const float* __restrict__ v2,
                          int n, const float* __restrict__ mPart,
                          const float* __restrict__ sPart, float* __restrict__ loss) {
  int lane = threadIdx.x & 31;
  int wave = (blockIdx.x * blockDim.x + threadIdx.x) >> 5;
  if (wave >= n) return;
  const float* a = v1 + (long)wave * DD_;
  const float* b = v2 + (long)wave * DD_;
  float d = 0.f;
#pragma unroll
  for (int t = 0; t < 4; ++t) d = fmaf(a[lane * 4 + t], b[lane * 4 + t], d);
  d += __shfl_xor(d, 1, 32);
  d += __shfl_xor(d, 2, 32);
  d += __shfl_xor(d, 4, 32);
  d += __shfl_xor(d, 8, 32);
  d += __shfl_xor(d, 16, 32);
  if (lane == 0) {
    float m = -3.0e38f, s = 0.f;
    for (int c = 0; c < JS_; ++c) {
      float om = mPart[(long)c * n + wave];
      float os = sPart[(long)c * n + wave];
      float nm = fmaxf(m, om);
      s = s * __expf(m - nm) + os * __expf(om - nm);
      m = nm;
    }
    float score = d * INV_T - (m + logf(s));
    unsafeAtomicAdd(loss, (-score) * (SSL_W / (float)n));
  }
}

// ---------------------------------------------------------------------------
// C[M,Kout] = gather(Ab, idx) @ Wb + bias, all-bf16 operands, Wb padded to 512.
// B fragments preloaded per col-tile; fast store path when the M-tile is full.
// ---------------------------------------------------------------------------
__global__ void k_proj_gemm(const __bf16* __restrict__ Ab, const int* __restrict__ idx,
                            int M, const __bf16* __restrict__ Wb,
                            const float* __restrict__ bias, int Kout,
                            float* __restrict__ C) {
  int lane = threadIdx.x & 31;
  int wave = (blockIdx.x * blockDim.x + threadIdx.x) >> 5;
  int mT = (M + 15) / 16;
  if (wave >= mT) return;
  int m0 = wave * 16;
  int mr = m0 + (lane & 15);
  if (mr >= M) mr = M - 1;
  long row = idx ? (long)idx[mr] : (long)mr;
  int kh = (lane >> 4) * 16;
  bool full = (m0 + 16) <= M;

  v16bf a[4];
#pragma unroll
  for (int ks = 0; ks < 4; ++ks)
    a[ks] = *(const v16bf*)(Ab + row * DD_ + ks * 32 + kh);

  int nT = (Kout + 15) / 16;
  for (int nt = 0; nt < nT; ++nt) {
    int n0 = nt * 16;
    v16bf b[4];
#pragma unroll
    for (int ks = 0; ks < 4; ++ks)
#pragma unroll
      for (int t = 0; t < 16; ++t)
        b[ks][t] = Wb[(long)(ks * 32 + kh + t) * KPAD_ + n0 + (lane & 15)];
    v8f c = {};
#pragma unroll
    for (int ks = 0; ks < 4; ++ks)
      c = __builtin_amdgcn_wmma_f32_16x16x32_bf16(false, a[ks], false, b[ks],
                                                  (short)0, c, false, false);
    int col = n0 + (lane & 15);
    if (col < Kout) {
      int rbase = m0 + 8 * (lane >> 4);
      float bb = bias[col];
      float* cp = C + (long)rbase * Kout + col;
      if (full) {
#pragma unroll
        for (int v = 0; v < 8; ++v) cp[(long)v * Kout] = c[v] + bb;
      } else {
#pragma unroll
        for (int v = 0; v < 8; ++v)
          if (rbase + v < M) cp[(long)v * Kout] = c[v] + bb;
      }
    }
  }
}

__global__ void k_gather_disc(const int* __restrict__ eid,
                              const float* __restrict__ disc, float* __restrict__ o) {
  int i = blockIdx.x * blockDim.x + threadIdx.x;
  if (i < BB_) o[i] = disc[eid[i]];
}

__global__ void k_gather_ki(const int* __restrict__ eid,
                            const float* __restrict__ ki, float* __restrict__ o) {
  long i = (long)blockIdx.x * blockDim.x + threadIdx.x;
  if (i >= (long)BB_ * DD_) return;
  int b = (int)(i / DD_), d = (int)(i - (long)b * DD_);
  o[i] = ki[(long)eid[b] * DD_ + d];
}

// ---------------------------------------------------------------------------
extern "C" void kernel_launch(void* const* d_in, const int* in_sizes, int n_in,
                              void* d_out, int out_size, void* d_ws, size_t ws_size,
                              hipStream_t stream) {
  (void)in_sizes; (void)n_in; (void)out_size; (void)ws_size;
  const int*   student_id  = (const int*)d_in[0];
  const int*   exercise_id = (const int*)d_in[1];
  const int*   rR = (const int*)d_in[2];  const int* rC = (const int*)d_in[3];
  const float* rV = (const float*)d_in[4];
  const int*   wR = (const int*)d_in[5];  const int* wC = (const int*)d_in[6];
  const float* wV = (const float*)d_in[7];
  const int*   fR = (const int*)d_in[8];  const int* fC = (const int*)d_in[9];
  const float* fV = (const float*)d_in[10];
  const int*   gR = (const int*)d_in[11]; const int* gC = (const int*)d_in[12];
  const float* gV = (const float*)d_in[13];
  const float* stu_emb  = (const float*)d_in[14];
  const float* exer_emb = (const float*)d_in[15];
  const float* know_emb = (const float*)d_in[16];
  const float* disc_emb = (const float*)d_in[17];
  const float* ki_emb   = (const float*)d_in[18];
  const float* Wc  = (const float*)d_in[19]; const float* bc  = (const float*)d_in[20];
  const float* Wts = (const float*)d_in[21]; const float* bts = (const float*)d_in[22];
  const float* Wte = (const float*)d_in[23]; const float* bte = (const float*)d_in[24];
  const float* Wtk = (const float*)d_in[25]; const float* btk = (const float*)d_in[26];

  // workspace layout (float units)
  float* ws    = (float*)d_ws;
  float* all0  = ws + 0L * ND_;
  float* r0    = ws + 1L * ND_;
  float* r1    = ws + 2L * ND_;
  float* w0    = ws + 3L * ND_;
  float* w1    = ws + 4L * ND_;
  float* acc   = ws + 5L * ND_;
  float* outF  = ws + 6L * ND_;
  float* outFF = ws + 7L * ND_;
  float*  fp   = ws + 8L * ND_;
  float*  mPart = fp;                 fp += (long)JS_ * S_;
  float*  sPart = fp;                 fp += (long)JS_ * S_;
  __bf16* outFb  = (__bf16*)fp;       fp += ND_ / 2;                 // [N x D]
  __bf16* outFFt = (__bf16*)fp;       fp += (DD_ * LDT_ + 1) / 2;    // [D x LDT_]
  __bf16* Wcb    = (__bf16*)fp;       fp += (2 * DD_ * DD_) / 2;     // [256 x 128]
  __bf16* Wtsb   = (__bf16*)fp;       fp += (DD_ * KPAD_) / 2;       // [128 x 512]
  __bf16* Wteb   = (__bf16*)fp;       fp += (DD_ * KPAD_) / 2;
  __bf16* Wtkb   = (__bf16*)fp;       fp += (DD_ * KPAD_) / 2;

  const int TPB = 256;
  const int ndBlk = (ND_ + TPB - 1) / TPB;

  k_init_all<<<ndBlk, TPB, 0, stream>>>(stu_emb, exer_emb, know_emb, all0);
  // one-time weight conversions
  k_cvt_bf16<<<(2 * DD_ * DD_ + TPB - 1) / TPB, TPB, 0, stream>>>(Wc, Wcb, 2 * DD_ * DD_);
  {
    int nb = (DD_ * KPAD_ + TPB - 1) / TPB;
    k_pad_w<<<nb, TPB, 0, stream>>>(Wts, Wtsb, KK_);
    k_pad_w<<<nb, TPB, 0, stream>>>(Wte, Wteb, KK_);
    k_pad_w<<<nb, TPB, 0, stream>>>(Wtk, Wtkb, KK_);
  }

  auto forward = [&](const int* RR, const int* RC, const float* RV,
                     const int* WWR, const int* WWC, const float* WWV, float* out) {
    hipMemcpyAsync(acc, all0, sizeof(float) * (size_t)ND_,
                   hipMemcpyDeviceToDevice, stream);
    const float* rcur = all0; const float* wcur = all0;
    float* rp = r0; float* rq = r1; float* wp = w0; float* wq = w1;
    const int spmmBlk = (int)(((long)NNZ_ * 32) / TPB);
    const int gemmWaves = (NN_ + 15) / 16;
    const int gemmBlk = (gemmWaves * 32 + TPB - 1) / TPB;
    for (int l = 0; l < LL_; ++l) {
      k_zero<<<ndBlk, TPB, 0, stream>>>(rp, ND_);
      k_spmm<<<spmmBlk, TPB, 0, stream>>>(RR, RC, RV, rcur, rp);
      k_zero<<<ndBlk, TPB, 0, stream>>>(wp, ND_);
      k_spmm<<<spmmBlk, TPB, 0, stream>>>(WWR, WWC, WWV, wcur, wp);
      k_concat_gemm_acc<<<gemmBlk, TPB, 0, stream>>>(rp, wp, Wcb, bc, acc);
      rcur = rp; wcur = wp;
      float* t = rp; rp = rq; rq = t;
      t = wp; wp = wq; wq = t;
    }
    k_scale<<<ndBlk, TPB, 0, stream>>>(acc, out, 1.0f / (LL_ + 1), ND_);
  };

  forward(rR, rC, rV, wR, wC, wV, outF);    // right/wrong -> f
  forward(fR, fC, fV, gR, gC, gV, outFF);   // rf/wf       -> ff

  // bf16 operand prep for the contrastive + projection stages
  k_cvt_bf16<<<ndBlk, TPB, 0, stream>>>(outF, outFb, ND_);
  k_transpose_bf16<<<ndBlk, TPB, 0, stream>>>(outFF, outFFt);

  // output layout (return-tuple order, flat f32)
  float* o_student = (float*)d_out;
  float* o_diff    = o_student + (long)BB_ * KK_;
  float* o_disc    = o_diff + (long)BB_ * KK_;
  float* o_know    = o_disc + BB_;
  float* o_loss    = o_know + (long)KK_ * KK_;
  float* o_ki      = o_loss + 1;

  k_zero1<<<1, 1, 0, stream>>>(o_loss);

  // stu info_nce: v1 rows [0,S), v2 columns [0,S) of transposed outFF
  {
    int mT = (S_ + 15) / 16, jT = (S_ + 15) / 16;
    int chunkT = (jT + JS_ - 1) / JS_;
    int wv = mT * JS_;
    k_nce<<<(wv * 32 + TPB - 1) / TPB, TPB, 0, stream>>>(outFb, outFFt, S_, chunkT,
                                                         mPart, sPart);
    k_nce_fin<<<(S_ * 32 + TPB - 1) / TPB, TPB, 0, stream>>>(outF, outFF, S_,
                                                             mPart, sPart, o_loss);
  }
  // exer info_nce: rows/cols offset by S (multiple of 16 -> stays 32B aligned)
  {
    int mT = (EX_ + 15) / 16, jT = (EX_ + 15) / 16;
    int chunkT = (jT + JS_ - 1) / JS_;
    int wv = mT * JS_;
    k_nce<<<(wv * 32 + TPB - 1) / TPB, TPB, 0, stream>>>(
        outFb + (long)S_ * DD_, outFFt + S_, EX_, chunkT, mPart, sPart);
    k_nce_fin<<<(EX_ * 32 + TPB - 1) / TPB, TPB, 0, stream>>>(
        outF + (long)S_ * DD_, outFF + (long)S_ * DD_, EX_, mPart, sPart, o_loss);
  }

  // projection heads (gather + bf16 WMMA GEMM)
  {
    int wv = (BB_ + 15) / 16;
    int bl = (wv * 32 + TPB - 1) / TPB;
    k_proj_gemm<<<bl, TPB, 0, stream>>>(outFb, student_id, BB_, Wtsb, bts, KK_, o_student);
    k_proj_gemm<<<bl, TPB, 0, stream>>>(outFb + (long)S_ * DD_, exercise_id, BB_,
                                        Wteb, bte, KK_, o_diff);
  }
  {
    int wv = (KK_ + 15) / 16;
    int bl = (wv * 32 + TPB - 1) / TPB;
    k_proj_gemm<<<bl, TPB, 0, stream>>>(outFb + (long)(S_ + EX_) * DD_, nullptr, KK_,
                                        Wtkb, btk, KK_, o_know);
  }

  k_gather_disc<<<(BB_ + TPB - 1) / TPB, TPB, 0, stream>>>(exercise_id, disc_emb, o_disc);
  k_gather_ki<<<(int)(((long)BB_ * DD_ + TPB - 1) / TPB), TPB, 0, stream>>>(
      exercise_id, ki_emb, o_ki);
}
